// CausalMultiHeadAttention_76897094468119
// MI455X (gfx1250) — matrix operand order repaired and checked by reading, not compiled, not run on hardware
//
#include <hip/hip_runtime.h>
#include <hip/hip_bf16.h>

// ---------------------------------------------------------------------------
// Causal MHA for MI455X (gfx1250, wave32, WMMA, TDM).
// Pipeline: f32->bf16 convert -> 3x WMMA GEMM (QKV proj) -> flash-attention
// (v_wmma_f32_16x16x32_bf16, K tiles staged via tensor_load_to_lds with
// double buffering) -> WMMA GEMM (output proj, fp32 out).
// ---------------------------------------------------------------------------

#define D_MODEL 1024
#define N_HEADS 16
#define D_KH    64
#define BATCH   4
#define SEQ     2048
#define M_ROWS  (BATCH * SEQ)   // 8192

typedef __attribute__((ext_vector_type(16))) __bf16 v16bf;
typedef __attribute__((ext_vector_type(8)))  float  v8f;
typedef __attribute__((ext_vector_type(4)))  unsigned int v4u;
typedef __attribute__((ext_vector_type(8)))  int v8i;
typedef __attribute__((ext_vector_type(4)))  int v4i;

#if defined(__AMDGCN__) && __has_builtin(__builtin_amdgcn_tensor_load_to_lds) && \
    __has_builtin(__builtin_amdgcn_s_wait_tensorcnt)
#define HAVE_TDM 1
#else
#define HAVE_TDM 0
#endif

union FragA { uint4 u[2]; v16bf v; };

__device__ __forceinline__ unsigned short f2bf(float f) {
  unsigned int u = __float_as_uint(f);
  u += 0x7FFFu + ((u >> 16) & 1u);       // round-to-nearest-even
  return (unsigned short)(u >> 16);
}

// ---------------------------------------------------------------------------
__global__ void cvt_f32_bf16(const float* __restrict__ in,
                             unsigned short* __restrict__ out, int n) {
  int i = blockIdx.x * blockDim.x + threadIdx.x;
  if (i < n) out[i] = f2bf(in[i]);
}

// ---------------------------------------------------------------------------
// C(M x N) = A(M x K) * W(N x K)^T + bias.  A, W are bf16 row-major.
// Block: 128 threads = 4 waves, block tile 128x128, each wave 64x64
// (4x4 grid of 16x16 f32 accumulators, K stepped by 32).
// mode 0: write bf16 into (B, H, S, 64) layout (for Q/K/V)
// mode 1: write fp32 row-major (final output)
// ---------------------------------------------------------------------------
__global__ __launch_bounds__(128)
void gemm_wmma(const unsigned short* __restrict__ A,
               const unsigned short* __restrict__ W,
               const float* __restrict__ bias,
               unsigned short* __restrict__ out_bf,
               float* __restrict__ out_f32,
               int K, int N, int mode) {
  const int lane = threadIdx.x & 31;
  const int wave = threadIdx.x >> 5;
  const int ml   = lane & 15;    // row (A) / col (B,C) selector
  const int hi   = lane >> 4;    // half-wave selector

  const int m0 = blockIdx.x * 128 + (wave >> 1) * 64;
  const int n0 = blockIdx.y * 128 + (wave & 1) * 64;

  v8f acc[4][4] = {};

  for (int k = 0; k < K; k += 32) {
    FragA a[4], b[4];
#pragma unroll
    for (int mt = 0; mt < 4; ++mt) {
      // A layout: lane ml holds row, 16 contiguous K at 16*hi
      const uint4* p = (const uint4*)(A + (size_t)(m0 + mt * 16 + ml) * K + k);
      a[mt].u[0] = p[2 * hi];
      a[mt].u[1] = p[2 * hi + 1];
    }
#pragma unroll
    for (int nt = 0; nt < 4; ++nt) {
      // B striping: lane ml = output column; 16 contiguous K at 16*hi
      const uint4* p = (const uint4*)(W + (size_t)(n0 + nt * 16 + ml) * K + k + 16 * hi);
      b[nt].u[0] = p[0];
      b[nt].u[1] = p[1];
    }
#pragma unroll
    for (int mt = 0; mt < 4; ++mt)
#pragma unroll
      for (int nt = 0; nt < 4; ++nt)
        acc[mt][nt] = __builtin_amdgcn_wmma_f32_16x16x32_bf16(
            false, a[mt].v, false, b[nt].v, (short)0, acc[mt][nt], false, false);
  }

#pragma unroll
  for (int nt = 0; nt < 4; ++nt) {
    const int gcol = n0 + nt * 16 + ml;
    const float bv = bias[gcol];
#pragma unroll
    for (int mt = 0; mt < 4; ++mt) {
#pragma unroll
      for (int r = 0; r < 8; ++r) {
        const int grow = m0 + mt * 16 + r + 8 * hi;   // C layout: M = r + 8*hi
        const float v = acc[mt][nt][r] + bv;
        if (mode == 1) {
          out_f32[(size_t)grow * N + gcol] = v;
        } else {
          const int bidx = grow >> 11;          // / SEQ
          const int s    = grow & (SEQ - 1);
          const int h    = gcol >> 6;           // / 64
          const int d    = gcol & 63;
          out_bf[(((size_t)bidx * N_HEADS + h) * SEQ + s) * D_KH + d] = f2bf(v);
        }
      }
    }
  }
}

#if HAVE_TDM
// ---------------------------------------------------------------------------
// TDM: DMA a 32(rows) x 64(features) bf16 K-tile (row-major, contiguous rows
// of 128 B) from global to LDS.  2D tensor: dim0 = 64 elems (contiguous),
// dim1 = SEQ rows, stride0 = 64, stride1 = 64*SEQ.  Tile: 64 x 32.
// Descriptor per CDNA5 ISA section 8 (groups 2/3 zero => <=2D tensor).
// ---------------------------------------------------------------------------
__device__ __forceinline__ void tdm_load_k_tile(const unsigned short* gsrc,
                                                unsigned int lds_byte_off) {
  const unsigned long long ga = (unsigned long long)(size_t)gsrc;
  v4u g0;
  g0.x = 1u;                                           // count=1, user D#
  g0.y = lds_byte_off;                                 // lds_addr
  g0.z = (unsigned int)(ga & 0xFFFFFFFFu);             // global_addr[31:0]
  g0.w = (unsigned int)((ga >> 32) & 0x01FFFFFFu)      // global_addr[56:32]
       | 0x80000000u;                                  // type=2 ("image")
  v8i g1;
  g1[0] = 0x00010000;                  // wg_mask=0, data_size=1 (2 bytes)
  g1[1] = (int)(64u << 16);            // tensor_dim0 = 64 (low16 @ [31:16])
  g1[2] = (int)((unsigned)SEQ << 16);  // dim0 hi16=0 | tensor_dim1 low16
  g1[3] = (int)(64u << 16);            // dim1 hi16=0 | tile_dim0 = 64
  g1[4] = 32;                          // tile_dim1 = 32, tile_dim2 = 0
  g1[5] = 64;                          // tensor_dim0_stride low32 = 64
  g1[6] = (int)(((64u * SEQ) & 0xFFFFu) << 16);  // stride0 hi16=0 | stride1 low16
  g1[7] = (int)((64u * SEQ) >> 16);              // tensor_dim1_stride high32
  const v4i z4 = {0, 0, 0, 0};
  const v8i z8 = {0, 0, 0, 0, 0, 0, 0, 0};
  // amdgpu-toolchain (clang-23) 6-arg form: (g0, g1, g2, g3, g4, cpol)
  __builtin_amdgcn_tensor_load_to_lds(g0, g1, z4, z4, z8, 0);
}
#endif

// ---------------------------------------------------------------------------
// Flash attention, causal + key padding mask.
// Grid: (SEQ/64, BATCH*N_HEADS). Block: 128 threads = 4 waves.
// Each wave owns 16 query rows; 32-key K tiles staged by the Tensor Data
// Mover into a double-buffered LDS tile (wave 0 issues, TENSORcnt waits);
// V tiles transposed at LDS-write time. Online softmax in fp32.
// ---------------------------------------------------------------------------
__global__ __launch_bounds__(128)
void flash_attn_wmma(const unsigned short* __restrict__ Q,
                     const unsigned short* __restrict__ Kmat,
                     const unsigned short* __restrict__ V,
                     const unsigned char* __restrict__ msk,
                     unsigned short* __restrict__ Obf) {
  __shared__ __align__(128) unsigned short lds_k2[2][32 * 64]; // K tiles (dbl buf)
  __shared__ __align__(16) unsigned short lds_vt[64 * 32];     // V tile, transposed
  __shared__ __align__(16) unsigned short lds_p[4][16 * 32];   // per-wave P tile

  const int lane = threadIdx.x & 31;
  const int wave = threadIdx.x >> 5;
  const int ml   = lane & 15;
  const int hi   = lane >> 4;

  const int bh = blockIdx.y;          // b * 16 + h
  const int bb = bh >> 4;
  const int q0 = blockIdx.x * 64;     // block's first query row
  const int qw = q0 + wave * 16;      // wave's first query row

  const unsigned short* Qp = Q    + (size_t)bh * SEQ * D_KH;
  const unsigned short* Kp = Kmat + (size_t)bh * SEQ * D_KH;
  const unsigned short* Vp = V    + (size_t)bh * SEQ * D_KH;

  // Q A-fragments for K-chunks [0,32) and [32,64): resident for whole kernel
  FragA aq[2];
#pragma unroll
  for (int c = 0; c < 2; ++c) {
    const uint4* p = (const uint4*)(Qp + (size_t)(qw + ml) * D_KH + 32 * c);
    aq[c].u[0] = p[2 * hi];
    aq[c].u[1] = p[2 * hi + 1];
  }

  v8f o[4] = {};               // 16x64 fp32 output accumulators
  float mrow[8], srow[8];
#pragma unroll
  for (int r = 0; r < 8; ++r) { mrow[r] = -__builtin_inff(); srow[r] = 0.f; }

  const int kmax = q0 + 64;    // causal bound (exclusive), uniform per block

#if HAVE_TDM
  const unsigned int koff0 = (unsigned int)(size_t)(&lds_k2[0][0]);
  const unsigned int koff1 = (unsigned int)(size_t)(&lds_k2[1][0]);
  if (wave == 0) tdm_load_k_tile(Kp, koff0);           // prologue DMA, tile 0
#endif

  for (int kb = 0; kb < kmax; kb += 32) {
    const int cur = (kb >> 5) & 1;
    const unsigned short* lk = lds_k2[cur];

#if HAVE_TDM
    if (wave == 0) {
      const int kn = kb + 32;
      if (kn < kmax) {  // kick DMA for next tile, then wait for current only
        tdm_load_k_tile(Kp + (size_t)kn * D_KH, cur ? koff0 : koff1);
        __builtin_amdgcn_s_wait_tensorcnt(1);
      } else {
        __builtin_amdgcn_s_wait_tensorcnt(0);
      }
    }
#else
    { // manual K staging (32x64): each thread moves 2x16B
      const int row = threadIdx.x >> 3;
      const int c0  = (threadIdx.x & 7) * 8;
      *(uint4*)(&lds_k2[cur][row * 64 + c0]) =
          *(const uint4*)(Kp + (size_t)(kb + row) * D_KH + c0);
      *(uint4*)(&lds_k2[cur][(row + 16) * 64 + c0]) =
          *(const uint4*)(Kp + (size_t)(kb + row + 16) * D_KH + c0);
    }
#endif

    // ---- stage V^T (64x32): transpose at LDS-write time
    {
      const int row = threadIdx.x >> 2;        // key 0..31
      const int c0  = (threadIdx.x & 3) * 16;  // feature 0..63 step 16
      union { uint4 u[2]; unsigned short s[16]; } t;
      const uint4* p = (const uint4*)(Vp + (size_t)(kb + row) * D_KH + c0);
      t.u[0] = p[0]; t.u[1] = p[1];
#pragma unroll
      for (int i = 0; i < 16; ++i) lds_vt[(c0 + i) * 32 + row] = t.s[i];
    }
    if (kb + 32 < kmax)        // global_prefetch_b8 of next V tile
      __builtin_prefetch(Vp + (size_t)(kb + 32) * D_KH, 0, 0);
    __syncthreads();

    // ---- scores: S = Q(16x64) . K_tile^T, two 16-key halves
    v8f sacc[2];
#pragma unroll
    for (int h2 = 0; h2 < 2; ++h2) {
      FragA bk0, bk1;
      {
        const uint4* p = (const uint4*)(&lk[(16 * h2 + ml) * 64 + 16 * hi]);
        bk0.u[0] = p[0]; bk0.u[1] = p[1];
        const uint4* q = (const uint4*)(&lk[(16 * h2 + ml) * 64 + 32 + 16 * hi]);
        bk1.u[0] = q[0]; bk1.u[1] = q[1];
      }
      v8f s = {};
      s = __builtin_amdgcn_wmma_f32_16x16x32_bf16(false, aq[0].v, false, bk0.v,
                                                  (short)0, s, false, false);
      s = __builtin_amdgcn_wmma_f32_16x16x32_bf16(false, aq[1].v, false, bk1.v,
                                                  (short)0, s, false, false);
      sacc[h2] = s;
    }

    // ---- mask + online softmax (C layout: row M = r + 8*hi, col = ml)
    bool mv[2];
    mv[0] = msk[(size_t)bb * SEQ + kb + ml] != 0;
    mv[1] = msk[(size_t)bb * SEQ + kb + 16 + ml] != 0;

    float pe[2][8];
#pragma unroll
    for (int r = 0; r < 8; ++r) {
      const int qrow = qw + r + 8 * hi;
      float mloc = -__builtin_inff();
#pragma unroll
      for (int h2 = 0; h2 < 2; ++h2) {
        const int kcol = kb + 16 * h2 + ml;
        float sc = sacc[h2][r] * 0.125f;   // 1/sqrt(64)
        const bool ok = (kcol <= qrow) && mv[h2];
        sc = ok ? sc : -__builtin_inff();
        pe[h2][r] = sc;
        mloc = fmaxf(mloc, sc);
      }
      // row-max across the 16 lanes holding this row
#pragma unroll
      for (int off = 1; off < 16; off <<= 1)
        mloc = fmaxf(mloc, __shfl_xor(mloc, off, 32));
      const float mnew  = fmaxf(mrow[r], mloc);
      const float alpha = (mrow[r] == -__builtin_inff()) ? 0.f
                                                         : __expf(mrow[r] - mnew);
      float ssum = 0.f;
#pragma unroll
      for (int h2 = 0; h2 < 2; ++h2) {
        const float p = (pe[h2][r] == -__builtin_inff())
                            ? 0.f
                            : __expf(pe[h2][r] - mnew);
        pe[h2][r] = p;
        ssum += p;
      }
#pragma unroll
      for (int off = 1; off < 16; off <<= 1)
        ssum += __shfl_xor(ssum, off, 32);
      srow[r] = srow[r] * alpha + ssum;
      mrow[r] = mnew;
#pragma unroll
      for (int nt = 0; nt < 4; ++nt) o[nt][r] = o[nt][r] * alpha;
    }

    // ---- P: C layout -> A layout via per-wave LDS tile (bf16)
    unsigned short* lp = lds_p[wave];
#pragma unroll
    for (int h2 = 0; h2 < 2; ++h2)
#pragma unroll
      for (int r = 0; r < 8; ++r)
        lp[(r + 8 * hi) * 32 + 16 * h2 + ml] = f2bf(pe[h2][r]);
    __syncthreads();

    // ---- O += P(16x32) . V(32x64)
    FragA pf;
    {
      const uint4* p = (const uint4*)(&lp[ml * 32]);
      pf.u[0] = p[2 * hi]; pf.u[1] = p[2 * hi + 1];
    }
#pragma unroll
    for (int nt = 0; nt < 4; ++nt) {
      FragA vf;
      const uint4* p = (const uint4*)(&lds_vt[(nt * 16 + ml) * 32 + 16 * hi]);
      vf.u[0] = p[0]; vf.u[1] = p[1];
      o[nt] = __builtin_amdgcn_wmma_f32_16x16x32_bf16(
          false, pf.v, false, vf.v, (short)0, o[nt], false, false);
    }
    __syncthreads();   // protect lds_k2 / lds_vt before next staging
  }

  // ---- normalize and write (B, S, H*64) bf16 for the output projection
  const int hd = bh & 15;
#pragma unroll
  for (int nt = 0; nt < 4; ++nt) {
#pragma unroll
    for (int r = 0; r < 8; ++r) {
      const int qrow = qw + r + 8 * hi;
      const float val = o[nt][r] / srow[r];
      Obf[((size_t)bb * SEQ + qrow) * D_MODEL + hd * D_KH + nt * 16 + ml] =
          f2bf(val);
    }
  }
}

// ---------------------------------------------------------------------------
extern "C" void kernel_launch(void* const* d_in, const int* in_sizes, int n_in,
                              void* d_out, int out_size, void* d_ws, size_t ws_size,
                              hipStream_t stream) {
  (void)in_sizes; (void)n_in; (void)out_size; (void)ws_size;

  const float* x            = (const float*)d_in[0];
  const unsigned char* mask = (const unsigned char*)d_in[1];
  const float* w_q = (const float*)d_in[2];
  const float* b_q = (const float*)d_in[3];
  const float* w_k = (const float*)d_in[4];
  const float* b_k = (const float*)d_in[5];
  const float* w_v = (const float*)d_in[6];
  const float* b_v = (const float*)d_in[7];
  const float* w_o = (const float*)d_in[8];
  const float* b_o = (const float*)d_in[9];

  char* ws = (char*)d_ws;
  size_t off = 0;
  auto alloc = [&](size_t bytes) -> char* {
    char* p = ws + off;
    off += (bytes + 255) & ~(size_t)255;
    return p;
  };
  unsigned short* x_bf    = (unsigned short*)alloc((size_t)M_ROWS * D_MODEL * 2);
  unsigned short* wq_bf   = (unsigned short*)alloc((size_t)D_MODEL * D_MODEL * 2);
  unsigned short* wk_bf   = (unsigned short*)alloc((size_t)D_MODEL * D_MODEL * 2);
  unsigned short* wv_bf   = (unsigned short*)alloc((size_t)D_MODEL * D_MODEL * 2);
  unsigned short* wo_bf   = (unsigned short*)alloc((size_t)D_MODEL * D_MODEL * 2);
  unsigned short* q_bf    = (unsigned short*)alloc((size_t)M_ROWS * D_MODEL * 2);
  unsigned short* k_bf    = (unsigned short*)alloc((size_t)M_ROWS * D_MODEL * 2);
  unsigned short* v_bf    = (unsigned short*)alloc((size_t)M_ROWS * D_MODEL * 2);
  unsigned short* attn_bf = (unsigned short*)alloc((size_t)M_ROWS * D_MODEL * 2);

  const int nX = M_ROWS * D_MODEL;
  const int nW = D_MODEL * D_MODEL;
  cvt_f32_bf16<<<(nX + 255) / 256, 256, 0, stream>>>(x,   x_bf,  nX);
  cvt_f32_bf16<<<(nW + 255) / 256, 256, 0, stream>>>(w_q, wq_bf, nW);
  cvt_f32_bf16<<<(nW + 255) / 256, 256, 0, stream>>>(w_k, wk_bf, nW);
  cvt_f32_bf16<<<(nW + 255) / 256, 256, 0, stream>>>(w_v, wv_bf, nW);
  cvt_f32_bf16<<<(nW + 255) / 256, 256, 0, stream>>>(w_o, wo_bf, nW);

  dim3 gg(M_ROWS / 128, D_MODEL / 128);
  gemm_wmma<<<gg, 128, 0, stream>>>(x_bf, wq_bf, b_q, q_bf, nullptr,
                                    D_MODEL, D_MODEL, 0);
  gemm_wmma<<<gg, 128, 0, stream>>>(x_bf, wk_bf, b_k, k_bf, nullptr,
                                    D_MODEL, D_MODEL, 0);
  gemm_wmma<<<gg, 128, 0, stream>>>(x_bf, wv_bf, b_v, v_bf, nullptr,
                                    D_MODEL, D_MODEL, 0);

  dim3 ga(SEQ / 64, BATCH * N_HEADS);
  flash_attn_wmma<<<ga, 128, 0, stream>>>(q_bf, k_bf, v_bf, mask, attn_bf);

  gemm_wmma<<<gg, 128, 0, stream>>>(attn_bf, wo_bf, b_o, nullptr, (float*)d_out,
                                    D_MODEL, D_MODEL, 1);
}
